// DotProductAttention_86294482911441
// MI455X (gfx1250) — matrix operand order repaired and checked by reading, not compiled, not run on hardware
//
#include <hip/hip_runtime.h>

typedef __attribute__((ext_vector_type(16))) __bf16 bf16x16;
typedef __attribute__((ext_vector_type(8)))  __bf16 bf16x8;
typedef __attribute__((ext_vector_type(4)))  __bf16 bf16x4;
typedef __attribute__((ext_vector_type(8)))  float  f32x8;

#define D_DIM 128   // head dim (and value dim)
#define KT    64    // keys per tile (4 x 16-key WMMA subtiles)
#define QW    16    // query rows per wave
#define WAVES 8     // waves per block (wave32)

#define KROW  144   // padded K row (bf16): 288B, 32B-aligned, bank-spread
#define VROW  80    // padded Vt / P row (bf16): 160B, 32B-aligned, bank-spread

__global__ __launch_bounds__(256)
void attn_fwd_bf16_wmma(const float* __restrict__ Q,
                        const float* __restrict__ K,
                        const float* __restrict__ V,
                        const int*   __restrict__ vlen,
                        float* __restrict__ O,
                        int Nq, int Nk)
{
    __shared__ __bf16 Klds[KT][KROW];          // K tile, row-major [key][d]
    __shared__ __bf16 Vt[D_DIM][VROW];         // V tile, transposed [v][key]
    __shared__ __bf16 Plds[WAVES][QW][VROW];   // per-wave P scratch (C->A relayout)

    const int tid  = threadIdx.x;
    const int wave = tid >> 5;
    const int lane = tid & 31;
    const int lh   = lane >> 4;   // half-wave: 0 or 1
    const int ln   = lane & 15;

    const int qblocks = Nq / (WAVES * QW);
    const int b    = blockIdx.x / qblocks;
    const int q0   = (blockIdx.x % qblocks) * (WAVES * QW);
    const int qrow = q0 + wave * QW + ln;

    const float scale = 0.08838834764831845f; // 1/sqrt(128)
    const int vl = vlen[b];

    // Cooperative-load geometry: flat float4 index = tid + i*256 maps linearly to
    // the 64x128 tile, so global offsets are (tid*16B + i*4096B) and LDS store
    // addresses are linear in i -> single vaddr + immediate offsets.
    const int krow0 = tid >> 5;          // K/V tile row for i=0 (rows advance by 8)
    const int c4    = (tid & 31) * 4;    // column within row (constant over i)

    // ---- Load Q A-fragments (16x32 bf16 layout), 4 fragments over D=128 ----
    // Contiguous k order used here (kb matches): lanes 0-15 hold M=lane, K = 0..15; lanes 16-31: K = 16..31
    bf16x16 qa[4];
    {
        const float* qp = Q + ((size_t)b * Nq + qrow) * D_DIM;
#pragma unroll
        for (int dk = 0; dk < 4; ++dk) {
            const float* plo = qp + dk * 32 + lh * 16;
            const float* phi = plo + 8;
            float4 a0 = *(const float4*)(plo);
            float4 a1 = *(const float4*)(plo + 4);
            float4 b0 = *(const float4*)(phi);
            float4 b1 = *(const float4*)(phi + 4);
            bf16x16 a;
            a[0]=(__bf16)a0.x; a[1]=(__bf16)a0.y; a[2]=(__bf16)a0.z; a[3]=(__bf16)a0.w;
            a[4]=(__bf16)a1.x; a[5]=(__bf16)a1.y; a[6]=(__bf16)a1.z; a[7]=(__bf16)a1.w;
            a[8]=(__bf16)b0.x; a[9]=(__bf16)b0.y; a[10]=(__bf16)b0.z; a[11]=(__bf16)b0.w;
            a[12]=(__bf16)b1.x; a[13]=(__bf16)b1.y; a[14]=(__bf16)b1.z; a[15]=(__bf16)b1.w;
            qa[dk] = a;
        }
    }

    // ---- Flash accumulators ----
    f32x8 o_acc[8];
#pragma unroll
    for (int f = 0; f < 8; ++f) o_acc[f] = (f32x8){};
    float m_run[8], l_run[8];
#pragma unroll
    for (int j = 0; j < 8; ++j) { m_run[j] = -3.0e38f; l_run[j] = 0.0f; }

    const float* srck = K + (size_t)b * Nk * D_DIM + tid * 4;  // per-thread base
    const float* srcv = V + (size_t)b * Nk * D_DIM + tid * 4;
    const int ntiles = Nk / KT;

    for (int kt = 0; kt < ntiles; ++kt) {
        __syncthreads();
        // ---- Cooperative tile load: global f32 -> LDS bf16 (linear addressing) ----
        {
#pragma unroll
            for (int i = 0; i < 8; ++i) {
                float4 kv = *(const float4*)(srck + i * 1024);
                bf16x4 w;
                w[0]=(__bf16)kv.x; w[1]=(__bf16)kv.y; w[2]=(__bf16)kv.z; w[3]=(__bf16)kv.w;
                *(bf16x4*)&Klds[krow0 + i * 8][c4] = w;
                float4 vv = *(const float4*)(srcv + i * 1024);
                Vt[c4 + 0][krow0 + i * 8] = (__bf16)vv.x;
                Vt[c4 + 1][krow0 + i * 8] = (__bf16)vv.y;
                Vt[c4 + 2][krow0 + i * 8] = (__bf16)vv.z;
                Vt[c4 + 3][krow0 + i * 8] = (__bf16)vv.w;
            }
            if (kt + 1 < ntiles) { // warm GL2 for the next tile (global_prefetch_b8)
                __builtin_prefetch(srck + KT * D_DIM, 0, 0);
                __builtin_prefetch(srcv + KT * D_DIM, 0, 0);
            }
            srck += KT * D_DIM;
            srcv += KT * D_DIM;
        }
        __syncthreads();

        // ---- S = Q * K^T : four 16x16 f32 subtiles (keys nsub*16 + ln) ----
        f32x8 s[4];
#pragma unroll
        for (int n = 0; n < 4; ++n) s[n] = (f32x8){};
#pragma unroll
        for (int dk = 0; dk < 4; ++dk) {
            const int d0 = dk * 32 + lh * 16; // k order used here (matches qa): half-wave splits K-dim 0..15 / 16..31
            bf16x16 kb[4];
#pragma unroll
            for (int n = 0; n < 4; ++n)
                kb[n] = *(const bf16x16*)&Klds[n * 16 + ln][d0];
#pragma unroll
            for (int n = 0; n < 4; ++n)
                s[n] = __builtin_amdgcn_wmma_f32_16x16x32_bf16(false, qa[dk], false, kb[n],
                                                               (short)0, s[n], false, false);
        }

        // ---- Mask (ref semantics: masked score := 1e-8, then softmax) + online softmax ----
        bool msk[4];
#pragma unroll
        for (int n = 0; n < 4; ++n) msk[n] = (kt * KT + n * 16 + ln) >= vl;
#pragma unroll
        for (int j = 0; j < 8; ++j) {
            float a[4];
#pragma unroll
            for (int n = 0; n < 4; ++n) a[n] = msk[n] ? 1e-8f : s[n][j] * scale;
            // row max: lane-local over 4 subtiles, then across the 16-lane row group
            float t = fmaxf(fmaxf(a[0], a[1]), fmaxf(a[2], a[3]));
#pragma unroll
            for (int off = 8; off; off >>= 1) t = fmaxf(t, __shfl_xor(t, off, 32));
            float mn  = fmaxf(m_run[j], t);
            float fac = __expf(m_run[j] - mn);
            float p[4];
            float r = 0.0f;
#pragma unroll
            for (int n = 0; n < 4; ++n) { p[n] = __expf(a[n] - mn); r += p[n]; }
#pragma unroll
            for (int off = 8; off; off >>= 1) r += __shfl_xor(r, off, 32);
            l_run[j] = l_run[j] * fac + r;
            m_run[j] = mn;
#pragma unroll
            for (int f = 0; f < 8; ++f) o_acc[f][j] *= fac;  // rescale O row
            // stash P in C-layout -> LDS (row M = j + 8*lh, key col = nsub*16 + ln)
            const int M = j + 8 * lh;
#pragma unroll
            for (int n = 0; n < 4; ++n) Plds[wave][M][n * 16 + ln] = (__bf16)p[n];
        }

        // ---- Reload P as two 16x32 A-fragments (wave-private LDS, DS ops in-order) ----
        bf16x16 pa0, pa1;
        {
            bf16x8 l0 = *(const bf16x8*)&Plds[wave][ln][lh * 16];
            bf16x8 h0 = *(const bf16x8*)&Plds[wave][ln][8 + lh * 16];
            bf16x8 l1 = *(const bf16x8*)&Plds[wave][ln][32 + lh * 16];
            bf16x8 h1 = *(const bf16x8*)&Plds[wave][ln][40 + lh * 16];
#pragma unroll
            for (int i = 0; i < 8; ++i) {
                pa0[i] = l0[i]; pa0[8 + i] = h0[i];
                pa1[i] = l1[i]; pa1[8 + i] = h1[i];
            }
        }

        // ---- O += P * V : 16 WMMAs over the 128 value columns ----
#pragma unroll
        for (int f = 0; f < 8; ++f) {
            bf16x16 vb0 = *(const bf16x16*)&Vt[f * 16 + ln][lh * 16];
            bf16x16 vb1 = *(const bf16x16*)&Vt[f * 16 + ln][32 + lh * 16];
            o_acc[f] = __builtin_amdgcn_wmma_f32_16x16x32_bf16(false, pa0, false, vb0,
                                                               (short)0, o_acc[f], false, false);
            o_acc[f] = __builtin_amdgcn_wmma_f32_16x16x32_bf16(false, pa1, false, vb1,
                                                               (short)0, o_acc[f], false, false);
        }
    }

    // ---- Normalize and store ----
    float inv_l[8];
#pragma unroll
    for (int j = 0; j < 8; ++j) inv_l[j] = 1.0f / l_run[j];
    float* op = O + ((size_t)b * Nq + q0 + wave * QW) * D_DIM;
#pragma unroll
    for (int f = 0; f < 8; ++f) {
#pragma unroll
        for (int j = 0; j < 8; ++j) {
            int M = j + 8 * lh;
            op[M * D_DIM + f * 16 + ln] = o_acc[f][j] * inv_l[j];
        }
    }
}

extern "C" void kernel_launch(void* const* d_in, const int* in_sizes, int n_in,
                              void* d_out, int out_size, void* d_ws, size_t ws_size,
                              hipStream_t stream) {
    const float* Q  = (const float*)d_in[0];
    const float* K  = (const float*)d_in[1];
    const float* V  = (const float*)d_in[2];
    const int*   vl = (const int*)d_in[3];
    float* O = (float*)d_out;

    const int B  = in_sizes[3];
    const int Nq = in_sizes[0] / (B * D_DIM);
    const int Nk = in_sizes[1] / (B * D_DIM);
    const int qblocks = Nq / (WAVES * QW);

    dim3 grid(B * qblocks);
    dim3 block(256);
    hipLaunchKernelGGL(attn_fwd_bf16_wmma, grid, block, 0, stream, Q, K, V, vl, O, Nq, Nk);
}
